// IVDetectModel_11441792877174
// MI455X (gfx1250) — matrix operand order repaired and checked by reading, not compiled, not run on hardware
//
#include <hip/hip_runtime.h>

typedef __attribute__((ext_vector_type(16))) __bf16 v16bf;
typedef __attribute__((ext_vector_type(8)))  float  v8f;

#define FEAT   128
#define NCLS   2
#define NGRAPH 64

// Native f32 -> bf16 conversion (hardware cvt on gfx1250)
static __device__ __forceinline__ __bf16 f2bf(float f) { return (__bf16)f; }

static __device__ __forceinline__ float4 relu4(float4 v) {
  v.x = fmaxf(v.x, 0.0f); v.y = fmaxf(v.y, 0.0f);
  v.z = fmaxf(v.z, 0.0f); v.w = fmaxf(v.w, 0.0f);
  return v;
}

// ---------------- degree / normalization ----------------
__global__ void k_init_deg(float* deg, int n) {
  int i = blockIdx.x * blockDim.x + threadIdx.x;
  if (i < n) deg[i] = 1.0f;                     // self-loop contribution
}

__global__ void k_accum_deg(const long long* __restrict__ dst, float* deg, int e) {
  int i = blockIdx.x * blockDim.x + threadIdx.x;
  if (i < e) atomicAdd(&deg[(int)dst[i]], 1.0f);
}

__global__ void k_rsqrt(float* deg, int n) {
  int i = blockIdx.x * blockDim.x + threadIdx.x;
  if (i < n) deg[i] = rsqrtf(deg[i]);           // deg >= 1 always
}

// ---------------- W -> fragment-linear bf16 ----------------
// Wswz flat index: (((s*8 + nt)*32 + lane)*16 + i)
//   B tile layout (16-bit B 32x16): n = nt*16 + (lane&15), kg = lane>>4,
//   k = 32*s + kg*8 + (i&7) + (i>>3)*16   (lane holds K [kg*8..kg*8+7] U [16+kg*8..16+kg*8+7])
__global__ void k_w_swizzle(const float* __restrict__ W, __bf16* __restrict__ Wswz) {
  int idx = blockIdx.x * blockDim.x + threadIdx.x;
  if (idx >= FEAT * FEAT) return;
  int i    = idx & 15;
  int lane = (idx >> 4) & 31;
  int nt   = (idx >> 9) & 7;
  int s    = idx >> 12;
  int k = 32 * s + (lane >> 4) * 8 + (i & 7) + (i >> 3) * 16;
  int n = nt * 16 + (lane & 15);
  Wswz[idx] = f2bf(W[k * FEAT + n]);
}

// ---------------- GEMM: Out = (relu?)H @ W, 8 waves x (16 rows x 128 cols) ----------------
// W (swizzled bf16, 32KB) is staged in LDS once per block; B fragments come
// from the ds path so the vmem pipe is dedicated to streaming A rows.
__global__ void __launch_bounds__(256)
k_gemm_wmma(const float* __restrict__ H, const __bf16* __restrict__ Wswz,
            float* __restrict__ Out, int nrows, int relu_in) {
  __shared__ __bf16 sW[FEAT * FEAT];            // 32 KB of the 320 KB WGP LDS

  // cooperative 16B-granule copy: 2048 uint4 across 256 threads
  {
    const uint4* gsrc = (const uint4*)Wswz;
    uint4*       ldst = (uint4*)sW;
    #pragma unroll
    for (int t = 0; t < (FEAT * FEAT) / (8 * 256); ++t)
      ldst[t * 256 + threadIdx.x] = gsrc[t * 256 + threadIdx.x];
  }
  __syncthreads();

  const int lane = threadIdx.x & 31;
  const int wave = threadIdx.x >> 5;
  const int row0 = blockIdx.x * 128 + wave * 16;
  const int m  = lane & 15;
  const int kg = lane >> 4;

  int arow = row0 + m;                  // clamp (no branch) to keep EXEC full
  if (arow > nrows - 1) arow = nrows - 1;
  const float* hrow = H + (long)arow * FEAT;

  v8f acc[8] = {};

#pragma unroll
  for (int s = 0; s < 4; ++s) {
    // A fragment (16-bit A 16x32): lane holds K in [kb..kb+7] U [kb+16..kb+23]
    const int kb = s * 32 + kg * 8;
    float4 f0 = *(const float4*)(hrow + kb);
    float4 f1 = *(const float4*)(hrow + kb + 4);
    float4 f2 = *(const float4*)(hrow + kb + 16);
    float4 f3 = *(const float4*)(hrow + kb + 20);
    if (relu_in) { f0 = relu4(f0); f1 = relu4(f1); f2 = relu4(f2); f3 = relu4(f3); }
    v16bf a;
    a[0]  = f2bf(f0.x); a[1]  = f2bf(f0.y); a[2]  = f2bf(f0.z); a[3]  = f2bf(f0.w);
    a[4]  = f2bf(f1.x); a[5]  = f2bf(f1.y); a[6]  = f2bf(f1.z); a[7]  = f2bf(f1.w);
    a[8]  = f2bf(f2.x); a[9]  = f2bf(f2.y); a[10] = f2bf(f2.z); a[11] = f2bf(f2.w);
    a[12] = f2bf(f3.x); a[13] = f2bf(f3.y); a[14] = f2bf(f3.z); a[15] = f2bf(f3.w);

#pragma unroll
    for (int nt = 0; nt < 8; ++nt) {
      v16bf b = *(const v16bf*)(sW + (((s * 8 + nt) * 32 + lane) << 4));
      acc[nt] = __builtin_amdgcn_wmma_f32_16x16x32_bf16(
          false, a, false, b, (short)0, acc[nt], false, false);
    }
  }

  // C/D layout: lane n = lane&15; rows M = (lane>>4)*8 + r
  const int n  = lane & 15;
  const int mg = lane >> 4;
#pragma unroll
  for (int nt = 0; nt < 8; ++nt) {
#pragma unroll
    for (int r = 0; r < 8; ++r) {
      int row = row0 + mg * 8 + r;
      if (row < nrows) Out[(long)row * FEAT + nt * 16 + n] = acc[nt][r];
    }
  }
}

// ---------------- bias splat (accumulator init) ----------------
__global__ void k_bias_init(float* __restrict__ out, const float* __restrict__ b, long total) {
  long i = (long)blockIdx.x * blockDim.x + threadIdx.x;
  if (i < total) out[i] = b[i & (FEAT - 1)];
}

// ---------------- edge + self-loop scatter-add: out[dst] += hW[src]*norm ----------------
__global__ void __launch_bounds__(256)
k_aggregate(const float* __restrict__ hW, const long long* __restrict__ src,
            const long long* __restrict__ dst, const float* __restrict__ dinv,
            float* __restrict__ out, int e, int n) {
  const int lane = threadIdx.x & 31;
  long w = (long)blockIdx.x * (blockDim.x >> 5) + (threadIdx.x >> 5);
  if (w >= (long)e + n) return;
  int s, d; float wt;
  if (w < e) {
    s = (int)src[w]; d = (int)dst[w];
    wt = dinv[s] * dinv[d];
  } else {
    s = d = (int)(w - e);
    float di = dinv[s]; wt = di * di;
  }
  float4 v = ((const float4*)(hW + (long)s * FEAT))[lane];
  float* o = out + (long)d * FEAT + lane * 4;
  atomicAdd(o + 0, v.x * wt);
  atomicAdd(o + 1, v.y * wt);
  atomicAdd(o + 2, v.z * wt);
  atomicAdd(o + 3, v.w * wt);
}

// ---------------- global max pool ----------------
__global__ void k_pool_init(float* pooled) {
  int i = blockIdx.x * blockDim.x + threadIdx.x;
  if (i < NGRAPH * FEAT) pooled[i] = -3.402823466e38f;
}

__global__ void k_pool_max(const float* __restrict__ h, const long long* __restrict__ batch,
                           float* __restrict__ pooled, int nnodes) {
  long i = (long)blockIdx.x * blockDim.x + threadIdx.x;
  if (i >= (long)nnodes * FEAT) return;
  int node = (int)(i >> 7);
  int c    = (int)(i & (FEAT - 1));
  int g    = (int)batch[node];
  atomicMax(&pooled[g * FEAT + c], h[i]);
}

// ---------------- final tiny linear: out = pooled @ Wf + bf ----------------
__global__ void k_final(const float* __restrict__ pooled, const float* __restrict__ Wf,
                        const float* __restrict__ bf_, float* __restrict__ out) {
  int t = threadIdx.x;                 // 128 threads: (graph, class)
  int g = t >> 1, c = t & 1;
  float acc = bf_[c];
#pragma unroll 8
  for (int k = 0; k < FEAT; ++k) acc += pooled[g * FEAT + k] * Wf[k * NCLS + c];
  out[g * NCLS + c] = acc;
}

extern "C" void kernel_launch(void* const* d_in, const int* in_sizes, int n_in,
                              void* d_out, int out_size, void* d_ws, size_t ws_size,
                              hipStream_t stream) {
  const float*     x     = (const float*)d_in[0];
  const long long* edge  = (const long long*)d_in[1];
  const long long* batch = (const long long*)d_in[2];
  const float* W[3] = { (const float*)d_in[3], (const float*)d_in[5], (const float*)d_in[7] };
  const float* b[3] = { (const float*)d_in[4], (const float*)d_in[6], (const float*)d_in[8] };
  const float* Wf  = (const float*)d_in[9];
  const float* bf_ = (const float*)d_in[10];

  const int n_nodes = in_sizes[0] / FEAT;   // 100000
  const int n_edges = in_sizes[1] / 2;      // 1600000
  const long long* src = edge;
  const long long* dst = edge + n_edges;

  // workspace carve-up
  char* ws = (char*)d_ws;
  float*  bufA   = (float*)ws;  ws += (size_t)n_nodes * FEAT * sizeof(float);
  float*  bufB   = (float*)ws;  ws += (size_t)n_nodes * FEAT * sizeof(float);
  float*  dinv   = (float*)ws;  ws += (size_t)n_nodes * sizeof(float);
  __bf16* Wswz   = (__bf16*)ws; ws += (size_t)FEAT * FEAT * sizeof(__bf16);
  float*  pooled = (float*)ws;  ws += (size_t)NGRAPH * FEAT * sizeof(float);

  const long total = (long)n_nodes * FEAT;
  const long nwork = (long)n_edges + n_nodes;
  const int  gemm_blocks = (n_nodes + 127) / 128;
  const int  agg_blocks  = (int)((nwork * 32 + 255) / 256);

  // D^{-1/2}
  k_init_deg <<<(n_nodes + 255) / 256, 256, 0, stream>>>(dinv, n_nodes);
  k_accum_deg<<<(n_edges + 255) / 256, 256, 0, stream>>>(dst, dinv, n_edges);
  k_rsqrt    <<<(n_nodes + 255) / 256, 256, 0, stream>>>(dinv, n_nodes);

  const float* gemm_in = x;
  for (int layer = 0; layer < 3; ++layer) {
    k_w_swizzle<<<(FEAT * FEAT + 255) / 256, 256, 0, stream>>>(W[layer], Wswz);
    k_gemm_wmma<<<gemm_blocks, 256, 0, stream>>>(gemm_in, Wswz, bufA, n_nodes,
                                                 layer > 0 ? 1 : 0);
    k_bias_init<<<(int)((total + 255) / 256), 256, 0, stream>>>(bufB, b[layer], total);
    k_aggregate<<<agg_blocks, 256, 0, stream>>>(bufA, src, dst, dinv, bufB,
                                                n_edges, n_nodes);
    gemm_in = bufB;   // ReLU of this output is fused into the next GEMM's A-load
  }

  k_pool_init<<<(NGRAPH * FEAT + 255) / 256, 256, 0, stream>>>(pooled);
  k_pool_max <<<(int)((total + 255) / 256), 256, 0, stream>>>(bufB, batch, pooled, n_nodes);
  k_final    <<<1, NGRAPH * NCLS, 0, stream>>>(pooled, Wf, bf_, (float*)d_out);
}
